// ActorCriticGNN_83416854823702
// MI455X (gfx1250) — compile-verified, matrix operand order checked by reading, not run on hardware
//
#include <hip/hip_runtime.h>
#include <hip/hip_bf16.h>

typedef __attribute__((ext_vector_type(16))) _Float16 v16h;
typedef __attribute__((ext_vector_type(8)))  float    v8f;

// ---- ordered-uint encoding for float atomic max ----
__device__ __forceinline__ unsigned fenc(float f) {
  unsigned u = __float_as_uint(f);
  return (u >> 31) ? ~u : (u | 0x80000000u);
}
__device__ __forceinline__ float fdec(unsigned u) {
  return (u >> 31) ? __uint_as_float(u & 0x7FFFFFFFu) : __uint_as_float(~u);
}

__global__ void k_zero(float* __restrict__ p, long n) {
  long i = (long)blockIdx.x * blockDim.x + threadIdx.x;
  long st = (long)gridDim.x * blockDim.x;
  for (; i < n; i += st) p[i] = 0.0f;
}

// cnt[d] += 1 ; sums[d][0..1] += edge_attr[e][0..1]
__global__ void k_edge_count(const int* __restrict__ dst, const float* __restrict__ eattr,
                             float* __restrict__ cnt, float* __restrict__ sums, int E) {
  int st = gridDim.x * blockDim.x;
  for (int e = blockIdx.x * blockDim.x + threadIdx.x; e < E; e += st) {
    int d = dst[e];
    atomicAdd(&cnt[d], 1.0f);
    atomicAdd(&sums[2 * d + 0], eattr[2 * e + 0]);
    atomicAdd(&sums[2 * d + 1], eattr[2 * e + 1]);
  }
}

__global__ void k_loop_attr(const float* __restrict__ cnt, float* __restrict__ loop, int n) {
  int i = blockIdx.x * blockDim.x + threadIdx.x;
  if (i >= n) return;
  float c = fmaxf(cnt[i], 1.0f);
  loop[2 * i + 0] /= c;
  loop[2 * i + 1] /= c;
}

// Y[n,64] = X[n,K] @ W[K,64] + bias ; f16 WMMA, f32 accumulate.
// Block = 128 threads = 4 waves, covers 16 rows x 64 cols (wave w -> cols [16w,16w+16)).
// W (K x 64) and the 16 x K A-tile are staged into LDS as f16 once per block; the
// four waves then read their WMMA fragments from LDS (A reads are contiguous 16B).
template <int K>
__global__ void __launch_bounds__(128) k_gemm_wmma(const float* __restrict__ X,
                                                   const float* __restrict__ W,
                                                   const float* __restrict__ bias,
                                                   float* __restrict__ Y, int n) {
  const int NC = 64;
  __shared__ _Float16 sW[K * NC];
  __shared__ _Float16 sA[16 * K];

  int tid  = threadIdx.x;
  int row0 = blockIdx.x * 16;

  // stage W -> LDS (f16)
#pragma unroll
  for (int i = tid; i < K * NC; i += 128) sW[i] = (_Float16)W[i];
  // stage A tile -> LDS (f16), clamped rows for the remainder tile
#pragma unroll
  for (int i = tid; i < 16 * K; i += 128) {
    int r  = row0 + i / K;
    int rc = r < n ? r : n - 1;
    sA[i]  = (_Float16)X[(long)rc * K + (i % K)];
  }
  __syncthreads();

  int lane = tid & 31;
  int wv   = tid >> 5;
  int m    = lane & 15;
  int kb   = (lane >> 4) * 8;          // 16-bit A/B striping: half-wave K split
  int col  = wv * 16 + (lane & 15);

  v8f c;
  float bv = bias[col];
#pragma unroll
  for (int i = 0; i < 8; ++i) c[i] = bv;

#pragma unroll
  for (int k0 = 0; k0 < K; k0 += 32) {
    v16h a, b;
    const _Float16* ar = sA + m * K + k0;
#pragma unroll
    for (int e = 0; e < 8; ++e) {
      a[e]     = ar[kb + e];                          // K = k0 + kb + e
      a[e + 8] = ar[16 + kb + e];                     // K = k0 + 16 + kb + e
    }
#pragma unroll
    for (int e = 0; e < 8; ++e) {
      b[e]     = sW[(k0 + kb + e) * NC + col];
      b[e + 8] = sW[(k0 + 16 + kb + e) * NC + col];
    }
    c = __builtin_amdgcn_wmma_f32_16x16x32_f16(false, a, false, b, (short)0, c,
                                               false, false);
  }

  // C/D layout: VGPR v holds M = v + 8*(lane>=16)
  int rbase = row0 + (lane >> 4) * 8;
  if (row0 + 16 <= n) {                 // uniform fast path: no EXEC juggling
#pragma unroll
    for (int v = 0; v < 8; ++v) Y[(long)(rbase + v) * NC + col] = c[v];
  } else {
#pragma unroll
    for (int v = 0; v < 8; ++v) {
      int r = rbase + v;
      if (r < n) Y[(long)r * NC + col] = c[v];
    }
  }
}

// Pass 1: per-edge logits + atomic segment max (ordered-u32).
template <int H, int C>
__global__ void k_edge_logits(const int* __restrict__ src, const int* __restrict__ dst,
                              const float* __restrict__ eattr, const float* __restrict__ loop,
                              const float* __restrict__ We, const float* __restrict__ att,
                              const float* __restrict__ xl, const float* __restrict__ xr,
                              float* __restrict__ logits, unsigned* __restrict__ nmax,
                              int E, int Etot) {
  const int HC = H * C;
  int st = gridDim.x * blockDim.x;
  for (int e = blockIdx.x * blockDim.x + threadIdx.x; e < Etot; e += st) {
    int s, d; float a0, a1;
    if (e < E) { s = src[e]; d = dst[e]; a0 = eattr[2 * e]; a1 = eattr[2 * e + 1]; }
    else       { s = e - E;  d = s;      a0 = loop[2 * s];  a1 = loop[2 * s + 1]; }
    const float* xls = xl + (long)s * HC;
    const float* xrd = xr + (long)d * HC;
    float lg[H];
#pragma unroll
    for (int h = 0; h < H; ++h) lg[h] = 0.0f;
#pragma unroll
    for (int j = 0; j < HC; ++j) {
      float z = xls[j] + xrd[j] + a0 * We[j] + a1 * We[HC + j];
      z = z > 0.0f ? z : 0.2f * z;          // leaky_relu(., 0.2)
      lg[j / C] += z * att[j];
    }
#pragma unroll
    for (int h = 0; h < H; ++h) {
      logits[(long)e * H + h] = lg[h];
      atomicMax(&nmax[(long)d * H + h], fenc(lg[h]));
    }
  }
}

// Pass 2: p = exp(logit - max); accumulate unnormalized messages + denominator.
template <int H, int C>
__global__ void k_edge_aggregate(const int* __restrict__ src, const int* __restrict__ dst,
                                 const float* __restrict__ xl, const float* __restrict__ logits,
                                 const unsigned* __restrict__ nmax,
                                 float* __restrict__ denom, float* __restrict__ acc,
                                 int E, int Etot) {
  const int HC = H * C;
  int st = gridDim.x * blockDim.x;
  for (int e = blockIdx.x * blockDim.x + threadIdx.x; e < Etot; e += st) {
    int s, d;
    if (e < E) { s = src[e]; d = dst[e]; } else { s = e - E; d = s; }
    const float* xls = xl + (long)s * HC;
    float p[H];
#pragma unroll
    for (int h = 0; h < H; ++h) {
      p[h] = __expf(logits[(long)e * H + h] - fdec(nmax[(long)d * H + h]));
      atomicAdd(&denom[(long)d * H + h], p[h]);
    }
    float* accd = acc + (long)d * HC;
#pragma unroll
    for (int j = 0; j < HC; ++j)
      atomicAdd(&accd[j], xls[j] * p[j / C]);
  }
}

// out = relu(acc/denom + bias)
template <int H, int C>
__global__ void k_node_finalize(const float* __restrict__ acc, const float* __restrict__ denom,
                                const float* __restrict__ bias, float* __restrict__ out, int n) {
  const int HC = H * C;
  long i = (long)blockIdx.x * blockDim.x + threadIdx.x;
  if (i >= (long)n * HC) return;
  int node = (int)(i / HC), j = (int)(i % HC);
  float v = acc[i] / denom[(long)node * H + j / C] + bias[j];
  out[i] = v > 0.0f ? v : 0.0f;
}

// actor/critic heads: action_mean=tanh(h@Wa+ba) [N,2]; std=exp(log_std) [2]; value=h@Wc+bc [N,1]
__global__ void k_heads(const float* __restrict__ h, const float* __restrict__ wa,
                        const float* __restrict__ ba, const float* __restrict__ wc,
                        const float* __restrict__ bc, const float* __restrict__ lstd,
                        float* __restrict__ out, int n) {
  int i = blockIdx.x * blockDim.x + threadIdx.x;
  if (i >= n) return;
  const float* hr = h + (long)i * 64;
  float a0 = ba[0], a1 = ba[1], v = bc[0];
#pragma unroll
  for (int j = 0; j < 64; ++j) {
    float x = hr[j];
    a0 += x * wa[2 * j + 0];
    a1 += x * wa[2 * j + 1];
    v  += x * wc[j];
  }
  out[2 * i + 0] = tanhf(a0);
  out[2 * i + 1] = tanhf(a1);
  out[(long)2 * n + 2 + i] = v;
  if (i == 0) {
    out[2 * n + 0] = __expf(lstd[0]);
    out[2 * n + 1] = __expf(lstd[1]);
  }
}

extern "C" void kernel_launch(void* const* d_in, const int* in_sizes, int n_in,
                              void* d_out, int out_size, void* d_ws, size_t ws_size,
                              hipStream_t stream) {
  (void)n_in; (void)out_size; (void)ws_size;
  const float* x     = (const float*)d_in[0];
  const int*   eidx  = (const int*)  d_in[1];
  const float* eattr = (const float*)d_in[2];
  const float* w1_l  = (const float*)d_in[3];
  const float* b1_l  = (const float*)d_in[4];
  const float* w1_r  = (const float*)d_in[5];
  const float* b1_r  = (const float*)d_in[6];
  const float* w1_e  = (const float*)d_in[7];
  const float* att1  = (const float*)d_in[8];
  const float* bias1 = (const float*)d_in[9];
  const float* w2_l  = (const float*)d_in[10];
  const float* b2_l  = (const float*)d_in[11];
  const float* w2_r  = (const float*)d_in[12];
  const float* b2_r  = (const float*)d_in[13];
  const float* w2_e  = (const float*)d_in[14];
  const float* att2  = (const float*)d_in[15];
  const float* bias2 = (const float*)d_in[16];
  const float* w_a   = (const float*)d_in[17];
  const float* b_a   = (const float*)d_in[18];
  const float* w_c   = (const float*)d_in[19];
  const float* b_c   = (const float*)d_in[20];
  const float* lstd  = (const float*)d_in[21];

  const int N    = in_sizes[0] / 32;
  const int E    = in_sizes[1] / 2;
  const int Etot = E + N;
  const int* src = eidx;
  const int* dst = eidx + E;

  // workspace layout (floats)
  float* ws = (float*)d_ws;
  size_t o = 0;
  float*    cnt    = ws + o; o += (size_t)N;
  float*    loop   = ws + o; o += (size_t)2 * N;
  float*    xl     = ws + o; o += (size_t)64 * N;
  float*    xr     = ws + o; o += (size_t)64 * N;
  float*    logits = ws + o; o += (size_t)4 * Etot;
  unsigned* nmax   = (unsigned*)(ws + o); o += (size_t)4 * N;
  float*    denom  = ws + o; o += (size_t)4 * N;
  float*    acc    = ws + o; o += (size_t)64 * N;
  float*    h1     = ws + o; o += (size_t)64 * N;
  float*    h2     = ws + o; o += (size_t)64 * N;

  const int ZB = 2048, TB = 256;
  const int egrid = 4096;
  const int ggrid = (N + 15) / 16;

  // ---- self-loop edge attrs (fill_value='mean' of incoming) ----
  k_zero<<<ZB, TB, 0, stream>>>(cnt, (long)3 * N);  // cnt + loop (contiguous)
  k_edge_count<<<egrid, TB, 0, stream>>>(dst, eattr, cnt, loop, E);
  k_loop_attr<<<(N + TB - 1) / TB, TB, 0, stream>>>(cnt, loop, N);

  // ---- layer 1: GATv2(D=32 -> H1=4, C1=16) ----
  k_gemm_wmma<32><<<ggrid, 128, 0, stream>>>(x, w1_l, b1_l, xl, N);
  k_gemm_wmma<32><<<ggrid, 128, 0, stream>>>(x, w1_r, b1_r, xr, N);
  k_zero<<<ZB, TB, 0, stream>>>((float*)nmax, (long)(4 + 4 + 64) * N); // nmax+denom+acc
  k_edge_logits<4, 16><<<egrid, TB, 0, stream>>>(src, dst, eattr, loop, w1_e, att1,
                                                 xl, xr, logits, nmax, E, Etot);
  k_edge_aggregate<4, 16><<<egrid, TB, 0, stream>>>(src, dst, xl, logits, nmax,
                                                    denom, acc, E, Etot);
  k_node_finalize<4, 16><<<((long)N * 64 + TB - 1) / TB, TB, 0, stream>>>(acc, denom,
                                                                          bias1, h1, N);

  // ---- layer 2: GATv2(64 -> H2=2, C2=32) ----
  k_gemm_wmma<64><<<ggrid, 128, 0, stream>>>(h1, w2_l, b2_l, xl, N);
  k_gemm_wmma<64><<<ggrid, 128, 0, stream>>>(h1, w2_r, b2_r, xr, N);
  k_zero<<<ZB, TB, 0, stream>>>((float*)nmax, (long)(4 + 4 + 64) * N);
  k_edge_logits<2, 32><<<egrid, TB, 0, stream>>>(src, dst, eattr, loop, w2_e, att2,
                                                 xl, xr, logits, nmax, E, Etot);
  k_edge_aggregate<2, 32><<<egrid, TB, 0, stream>>>(src, dst, xl, logits, nmax,
                                                    denom, acc, E, Etot);
  k_node_finalize<2, 32><<<((long)N * 64 + TB - 1) / TB, TB, 0, stream>>>(acc, denom,
                                                                          bias2, h2, N);

  // ---- heads ----
  k_heads<<<(N + TB - 1) / TB, TB, 0, stream>>>(h2, w_a, b_a, w_c, b_c, lstd,
                                                (float*)d_out, N);
}
